// MambaTrackletPredictor_79577154060665
// MI455X (gfx1250) — matrix operand order repaired and checked by reading, not compiled, not run on hardware
//
#include <hip/hip_runtime.h>
#include <hip/hip_bf16.h>

#define BATCH   64
#define SEQ     512
#define D_MODEL 64
#define D_INNER 128
#define D_STATE 16
#define D_CONV  4
#define DT_RANK 4
#define MLP_H   64
#define TRAJ    64
#define BL      (BATCH * SEQ)   // 32768 rows

typedef __attribute__((ext_vector_type(16))) __bf16 v16bf;
typedef __attribute__((ext_vector_type(8)))  float  v8f;

// ---------------------------------------------------------------------------
// bf16 WMMA GEMM:  C[M,Nstride] = A[M,K] @ W[Nreal,K]^T (+bias)
// Template: K (64/128, fully unrolled), NT (n-tiles per wave), PAD
// (predicate-free zero-fill for N>Nreal).
//
// Per k-step: one A fragment + all NT B fragments are loaded into distinct
// registers first, then NT WMMAs issue back-to-back (disjoint D/A/B between
// consecutive WMMAs -> no hazard NOPs; the WMMA chain also covers the
// WMMA->VALU hazard window of the next address computation).
//
// A fragment (16x32 bf16, ISA 7.12.2): lane half 0 rows M=0..15 hold K
// k0+0..15 (elems 0..15); half 1 holds
// k0+16..31. B fragment (32x16): lane holds column n = lane%16, elems 0..15 =
// 16 consecutive K at k0 + 16*half. C/D: vgpr r, lane l -> row r+8*(l>=16),
// col l%16.
// ---------------------------------------------------------------------------
template<int K, int NT, bool PAD>
__global__ __launch_bounds__(128)
void wmma_gemm_bf16(const float* __restrict__ A,
                    const float* __restrict__ W,
                    const float* __restrict__ bias,
                    float* __restrict__ C,
                    int M, int N, int Nreal, int Nstride)
{
    const int wave = (blockIdx.x * blockDim.x + threadIdx.x) >> 5;
    const int lane = threadIdx.x & 31;
    const int ngN  = N / (16 * NT);                  // n-groups
    const int totalWaves = (M >> 4) * ngN;
    if (wave >= totalWaves) return;                  // wave-uniform exit

    const int gn = wave % ngN;
    const int tm = wave / ngN;
    const int half = lane >> 4;                      // 0 or 1
    const int l    = lane & 15;
    const int rowA = tm * 16 + l;

    v8f acc[NT];
    #pragma unroll
    for (int nt = 0; nt < NT; ++nt)
        #pragma unroll
        for (int i = 0; i < 8; ++i) acc[nt][i] = 0.0f;

    #pragma unroll
    for (int k0 = 0; k0 < K; k0 += 32) {
        // ---- A fragment: loaded once, feeds NT WMMAs ----
        v16bf afrag;
        const float* ap = A + (size_t)rowA * K + k0 + half * 16;
        #pragma unroll
        for (int j = 0; j < 8; ++j) afrag[j]     = (__bf16)ap[j];
        #pragma unroll
        for (int j = 0; j < 8; ++j) afrag[8 + j] = (__bf16)ap[8 + j];

        // ---- all NT B fragments into distinct registers ----
        v16bf bfrag[NT];
        #pragma unroll
        for (int nt = 0; nt < NT; ++nt) {
            const int colB = (gn * NT + nt) * 16 + l;
            if (PAD) {
                // branch-free zero-fill: clamp row, scale OOB lanes by 0
                int   colW   = colB < Nreal ? colB : (Nreal - 1);
                float wscale = colB < Nreal ? 1.0f : 0.0f;
                const float* bp = W + (size_t)colW * K + k0 + half * 16;
                #pragma unroll
                for (int j = 0; j < 16; ++j) bfrag[nt][j] = (__bf16)(bp[j] * wscale);
            } else {
                const float* bp = W + (size_t)colB * K + k0 + half * 16;
                #pragma unroll
                for (int j = 0; j < 16; ++j) bfrag[nt][j] = (__bf16)bp[j];
            }
        }

        // ---- NT back-to-back WMMAs ----
        #pragma unroll
        for (int nt = 0; nt < NT; ++nt)
            acc[nt] = __builtin_amdgcn_wmma_f32_16x16x32_bf16(
                          false, afrag, false, bfrag[nt], (short)0, acc[nt], false, false);
    }

    #pragma unroll
    for (int nt = 0; nt < NT; ++nt) {
        const int col = (gn * NT + nt) * 16 + l;
        float bv = 0.0f;
        if (bias != nullptr) bv = (!PAD || col < Nreal) ? bias[col] : 0.0f;
        #pragma unroll
        for (int r = 0; r < 8; ++r) {
            int row = tm * 16 + r + half * 8;
            C[(size_t)row * Nstride + col] = acc[nt][r] + bv;
        }
    }
}

// ---------------------------------------------------------------------------
// Encoder: h[bl, m] = bbox[bl, 0:4] . enc_w[m, 0:4] + enc_b[m]   (K=4, VALU)
// ---------------------------------------------------------------------------
__global__ void encoder_kernel(const float* __restrict__ bbox,
                               const float* __restrict__ enc_w,
                               const float* __restrict__ enc_b,
                               float* __restrict__ h)
{
    int idx = blockIdx.x * blockDim.x + threadIdx.x;
    if (idx >= BL * D_MODEL) return;
    int m  = idx & (D_MODEL - 1);
    int bl = idx >> 6;
    float acc = enc_b[m];
    #pragma unroll
    for (int k = 0; k < 4; ++k)
        acc += bbox[(size_t)bl * 4 + k] * enc_w[m * 4 + k];
    h[idx] = acc;
}

// ---------------------------------------------------------------------------
// Depthwise causal conv(4) + bias + SiLU.  xi = xz[:, 0:128].
// ---------------------------------------------------------------------------
__global__ void conv_silu_kernel(const float* __restrict__ xz,
                                 const float* __restrict__ conv_w,
                                 const float* __restrict__ conv_b,
                                 float* __restrict__ xc)
{
    int idx = blockIdx.x * blockDim.x + threadIdx.x;
    if (idx >= BL * D_INNER) return;
    int d  = idx & (D_INNER - 1);
    int bl = idx >> 7;
    int t  = bl & (SEQ - 1);
    int blBase = bl - t;                            // (b, t=0)
    float acc = conv_b[d];
    #pragma unroll
    for (int k = 0; k < D_CONV; ++k) {
        int tt = t - (D_CONV - 1) + k;
        if (tt >= 0)
            acc += xz[(size_t)(blBase + tt) * (2 * D_INNER) + d] * conv_w[d * 4 + k];
    }
    xc[idx] = acc / (1.0f + __expf(-acc));          // SiLU
}

// ---------------------------------------------------------------------------
// Fused: dt_proj + softplus + selective scan + C-reduce + D-skip + SiLU gate.
// grid = (BATCH, 4), block = 32 lanes; lane -> channel d = 32*gy + lane.
// 16 states per lane held in VGPRs; per-step dbc row staged via LDS.
// ---------------------------------------------------------------------------
__global__ __launch_bounds__(32)
void scan_kernel(const float* __restrict__ dbc,   // [BL, 48] (0:4 dt, 4:20 B, 20:36 C)
                 const float* __restrict__ xc,    // [BL, 128]
                 const float* __restrict__ xz,    // [BL, 256] (z = cols 128:256)
                 const float* __restrict__ dt_w,  // [128, 4]
                 const float* __restrict__ dt_b,  // [128]
                 const float* __restrict__ A_log, // [128, 16]
                 const float* __restrict__ Dp,    // [128]
                 float* __restrict__ y)           // [BL, 128]
{
    const int b = blockIdx.x;
    const int d = blockIdx.y * 32 + threadIdx.x;
    __shared__ float sh[40];

    float Arow[D_STATE];
    #pragma unroll
    for (int s = 0; s < D_STATE; ++s)
        Arow[s] = -__expf(A_log[d * D_STATE + s]);

    const float w0 = dt_w[d * 4 + 0], w1 = dt_w[d * 4 + 1];
    const float w2 = dt_w[d * 4 + 2], w3 = dt_w[d * 4 + 3];
    const float bdt = dt_b[d];
    const float Dd  = Dp[d];

    float h[D_STATE];
    #pragma unroll
    for (int s = 0; s < D_STATE; ++s) h[s] = 0.0f;

    for (int t = 0; t < SEQ; ++t) {
        const size_t bl = (size_t)b * SEQ + t;
        int ln = threadIdx.x;
        sh[ln] = dbc[bl * 48 + ln];                 // 0..31
        if (ln < 4) sh[32 + ln] = dbc[bl * 48 + 32 + ln];
        __syncthreads();

        float dtr = sh[0] * w0 + sh[1] * w1 + sh[2] * w2 + sh[3] * w3 + bdt;
        float dt  = (dtr > 20.0f) ? dtr : __logf(1.0f + __expf(dtr)); // softplus
        float x   = xc[bl * D_INNER + d];

        float yt = 0.0f;
        #pragma unroll
        for (int s = 0; s < D_STATE; ++s) {
            float dA = __expf(dt * Arow[s]);
            h[s] = dA * h[s] + dt * sh[4 + s] * x;  // B = sh[4..19]
            yt  += h[s] * sh[20 + s];               // C = sh[20..35]
        }
        yt += Dd * x;
        float z = xz[bl * (2 * D_INNER) + D_INNER + d];
        y[bl * D_INNER + d] = yt * (z / (1.0f + __expf(-z)));
        __syncthreads();
    }
}

// ---------------------------------------------------------------------------
// Heads: last timestep -> delta MLP (64->64->4) and traj MLP (64->64->64)+LN.
// grid = BATCH, block = 64.
// ---------------------------------------------------------------------------
__global__ __launch_bounds__(64)
void head_kernel(const float* __restrict__ h2,    // [BL, 64]
                 const float* __restrict__ w1d, const float* __restrict__ b1d,
                 const float* __restrict__ w2d, const float* __restrict__ b2d,
                 const float* __restrict__ w1t, const float* __restrict__ b1t,
                 const float* __restrict__ w2t, const float* __restrict__ b2t,
                 const float* __restrict__ ln_s, const float* __restrict__ ln_b,
                 float* __restrict__ out)         // [256] delta ++ [4096] traj
{
    const int b = blockIdx.x, j = threadIdx.x;
    __shared__ float last[64], hid[64], red[64];

    last[j] = h2[((size_t)b * SEQ + (SEQ - 1)) * D_MODEL + j];
    __syncthreads();

    // ---- delta MLP ----
    float acc = b1d[j];
    #pragma unroll 8
    for (int k = 0; k < 64; ++k) acc += last[k] * w1d[j * 64 + k];
    hid[j] = fmaxf(acc, 0.0f);
    __syncthreads();
    if (j < 4) {
        float o = b2d[j];
        #pragma unroll 8
        for (int k = 0; k < 64; ++k) o += hid[k] * w2d[j * 64 + k];
        out[b * 4 + j] = o;
    }
    __syncthreads();

    // ---- traj MLP ----
    acc = b1t[j];
    #pragma unroll 8
    for (int k = 0; k < 64; ++k) acc += last[k] * w1t[j * 64 + k];
    hid[j] = fmaxf(acc, 0.0f);
    __syncthreads();
    float o = b2t[j];
    #pragma unroll 8
    for (int k = 0; k < 64; ++k) o += hid[k] * w2t[j * 64 + k];

    // ---- LayerNorm over 64 outputs ----
    red[j] = o; __syncthreads();
    for (int st = 32; st > 0; st >>= 1) { if (j < st) red[j] += red[j + st]; __syncthreads(); }
    float mu = red[0] / 64.0f;
    __syncthreads();
    red[j] = (o - mu) * (o - mu); __syncthreads();
    for (int st = 32; st > 0; st >>= 1) { if (j < st) red[j] += red[j + st]; __syncthreads(); }
    float var = red[0] / 64.0f;
    out[BATCH * 4 + b * 64 + j] = (o - mu) * rsqrtf(var + 1e-5f) * ln_s[j] + ln_b[j];
}

// ---------------------------------------------------------------------------
// Host orchestration
// ---------------------------------------------------------------------------
static inline int gemm_blocks(int M, int N, int NT)
{
    int waves = (M / 16) * (N / (16 * NT));
    return (waves + 3) / 4;                          // 4 waves (128 thr) / block
}

static void run_mamba(const float* hin, float* hout,
                      const float* in_proj_w, const float* conv_w, const float* conv_b,
                      const float* x_proj_w, const float* dt_w, const float* dt_b,
                      const float* A_log, const float* Dp, const float* out_proj_w,
                      float* xz, float* xc, float* dbc, float* ybuf,
                      hipStream_t stream)
{
    // in_proj: [BL,64] @ [256,64]^T -> xz [BL,256]   (K=64, 16x64 strip/wave)
    wmma_gemm_bf16<64, 4, false><<<gemm_blocks(BL, 256, 4), 128, 0, stream>>>(
        hin, in_proj_w, nullptr, xz, BL, 2 * D_INNER, 2 * D_INNER, 2 * D_INNER);
    // conv + SiLU -> xc [BL,128]
    conv_silu_kernel<<<(BL * D_INNER + 255) / 256, 256, 0, stream>>>(xz, conv_w, conv_b, xc);
    // x_proj: [BL,128] @ [36,128]^T -> dbc [BL,48]   (N padded 36->48, 3 tiles/wave)
    wmma_gemm_bf16<128, 3, true><<<gemm_blocks(BL, 48, 3), 128, 0, stream>>>(
        xc, x_proj_w, nullptr, dbc, BL, 48, DT_RANK + 2 * D_STATE, 48);
    // fused dt_proj + softplus + scan + gate -> ybuf [BL,128]
    scan_kernel<<<dim3(BATCH, 4), 32, 0, stream>>>(dbc, xc, xz, dt_w, dt_b, A_log, Dp, ybuf);
    // out_proj: [BL,128] @ [64,128]^T -> hout [BL,64]  (4 tiles/wave)
    wmma_gemm_bf16<128, 4, false><<<gemm_blocks(BL, 64, 4), 128, 0, stream>>>(
        ybuf, out_proj_w, nullptr, hout, BL, D_MODEL, D_MODEL, D_MODEL);
}

extern "C" void kernel_launch(void* const* d_in, const int* in_sizes, int n_in,
                              void* d_out, int out_size, void* d_ws, size_t ws_size,
                              hipStream_t stream)
{
    // Input order: bbox, enc_w, enc_b,
    //              m1{in_proj_w,conv_w,conv_b,x_proj_w,dt_proj_w,dt_proj_b,A_log,D,out_proj_w},
    //              m2{...}, mlp_delta{w1,b1,w2,b2}, mlp_traj{w1,b1,w2,b2,ln_s,ln_b}
    const float* bbox  = (const float*)d_in[0];
    const float* enc_w = (const float*)d_in[1];
    const float* enc_b = (const float*)d_in[2];
    const float* m1[9]; for (int i = 0; i < 9; ++i) m1[i] = (const float*)d_in[3 + i];
    const float* m2[9]; for (int i = 0; i < 9; ++i) m2[i] = (const float*)d_in[12 + i];
    const float* w1d = (const float*)d_in[21], *b1d = (const float*)d_in[22];
    const float* w2d = (const float*)d_in[23], *b2d = (const float*)d_in[24];
    const float* w1t = (const float*)d_in[25], *b1t = (const float*)d_in[26];
    const float* w2t = (const float*)d_in[27], *b2t = (const float*)d_in[28];
    const float* lns = (const float*)d_in[29], *lnb = (const float*)d_in[30];

    // workspace layout (fp32): ~90 MB
    float* ws    = (float*)d_ws;
    float* hbuf  = ws;                         // BL*64
    float* xz    = hbuf + (size_t)BL * 64;     // BL*256
    float* xc    = xz   + (size_t)BL * 256;    // BL*128
    float* dbc   = xc   + (size_t)BL * 128;    // BL*48
    float* ybuf  = dbc  + (size_t)BL * 48;     // BL*128
    float* hbuf2 = ybuf + (size_t)BL * 128;    // BL*64

    // encoder
    encoder_kernel<<<(BL * D_MODEL + 255) / 256, 256, 0, stream>>>(bbox, enc_w, enc_b, hbuf);

    // mamba block 1: hbuf -> hbuf2
    run_mamba(hbuf, hbuf2, m1[0], m1[1], m1[2], m1[3], m1[4], m1[5], m1[6], m1[7], m1[8],
              xz, xc, dbc, ybuf, stream);
    // mamba block 2: hbuf2 -> hbuf
    run_mamba(hbuf2, hbuf, m2[0], m2[1], m2[2], m2[3], m2[4], m2[5], m2[6], m2[7], m2[8],
              xz, xc, dbc, ybuf, stream);

    // heads (last timestep)
    head_kernel<<<BATCH, 64, 0, stream>>>(hbuf, w1d, b1d, w2d, b2d,
                                          w1t, b1t, w2t, b2t, lns, lnb,
                                          (float*)d_out);
    (void)in_sizes; (void)n_in; (void)out_size; (void)ws_size;
}
